// _MoEMLP_66855460929597
// MI455X (gfx1250) — matrix operand order repaired and checked by reading, not compiled, not run on hardware
//
#include <hip/hip_runtime.h>
#include <hip/hip_bf16.h>

// MoE SwiGLU (B=4,T=2048,D=768,E=8,K=2,H=3072) for gfx1250 (MI455X).
// Pipeline: fp32->bf16 convert | router top-2 + gather lists | fused
// per-expert WMMA GEMM (gate/up -> silu*up -> down-proj) | combine slots.
// bf16 WMMA (v_wmma_f32_16x16x32_bf16) with f32 accumulation; weights
// converted once to bf16 (113 MB) so per-tile streaming hits the 192MB L2.
// x-tile gather staged with GLOBAL_LOAD_ASYNC_TO_LDS_B128 (ASYNCcnt path).
// Workspace use: ~177 MB.

#define N_TOK 8192
#define DDIM  768
#define EEXP  8
#define HDIM  3072
#define TM    32     // tokens per block tile
#define HC    64     // H-chunk per iteration

typedef __attribute__((ext_vector_type(16))) __bf16 v16bf;
typedef __attribute__((ext_vector_type(8)))  float  v8f;

union BfVec {
  uint4 q[2];
  v16bf v;
  unsigned short u[16];
};

__device__ __forceinline__ unsigned short f2bf(float f) {
  unsigned int u = __float_as_uint(f);
  unsigned int r = u + 0x7FFFu + ((u >> 16) & 1u);   // round-to-nearest-even
  return (unsigned short)(r >> 16);
}

// Async memory->LDS 16B copy (gfx1250 GLOBAL_LOAD_ASYNC_TO_LDS_B128).
// Generic LDS pointers carry the LDS offset in their low 32 bits (ISA 10.2),
// which is what the VDST VGPR of the async op expects.
__device__ __forceinline__ void async_copy_b128(void* lds_dst, const void* gsrc) {
#if defined(__gfx1250__)
  unsigned loff = (unsigned)(unsigned long long)lds_dst;
  asm volatile("global_load_async_to_lds_b128 %0, %1, off"
               :: "v"(loff), "v"(gsrc) : "memory");
#else
  *(uint4*)lds_dst = *(const uint4*)gsrc;
#endif
}

__device__ __forceinline__ void async_wait0() {
#if defined(__gfx1250__)
  asm volatile("s_wait_asynccnt 0x0" ::: "memory");
#endif
}

// ---------------------------------------------------------------- convert
__global__ void _moe_cvt_kernel(const float* __restrict__ x,
                                const float* __restrict__ wg,
                                const float* __restrict__ wu,
                                const float* __restrict__ wd,
                                unsigned short* __restrict__ xb,
                                unsigned short* __restrict__ wgb,
                                unsigned short* __restrict__ wub,
                                unsigned short* __restrict__ wdb) {
  const long long S0 = (long long)N_TOK * DDIM;
  const long long S1 = (long long)EEXP * HDIM * DDIM;
  const long long total4 = (S0 + 3 * S1) >> 2;
  for (long long i = blockIdx.x * (long long)blockDim.x + threadIdx.x;
       i < total4; i += (long long)gridDim.x * blockDim.x) {
    long long f = i << 2;
    const float* src; unsigned short* dst; long long off;
    if (f < S0)            { src = x;  dst = xb;  off = f; }
    else if (f < S0 + S1)  { src = wg; dst = wgb; off = f - S0; }
    else if (f < S0 + 2*S1){ src = wu; dst = wub; off = f - S0 - S1; }
    else                   { src = wd; dst = wdb; off = f - S0 - 2*S1; }
    float4 v = *(const float4*)(src + off);
    unsigned int lo = (unsigned int)f2bf(v.x) | ((unsigned int)f2bf(v.y) << 16);
    unsigned int hi = (unsigned int)f2bf(v.z) | ((unsigned int)f2bf(v.w) << 16);
    *(uint2*)(dst + off) = make_uint2(lo, hi);
  }
}

// ---------------------------------------------------------------- router
// one wave32 per token: 8 dot products over D=768, lane-split + shfl reduce,
// top-2 + softmax, atomic compaction into per-expert (token,slot,weight) lists.
__global__ void _moe_router_kernel(const float* __restrict__ x,
                                   const float* __restrict__ rw,
                                   int* __restrict__ cnt,
                                   int* __restrict__ tokl,
                                   float* __restrict__ wl) {
  const int wave = threadIdx.x >> 5;
  const int lane = threadIdx.x & 31;
  const int t = blockIdx.x * (blockDim.x >> 5) + wave;
  if (t >= N_TOK) return;
  float acc[EEXP];
  #pragma unroll
  for (int e = 0; e < EEXP; ++e) acc[e] = 0.f;
  const float* xr = x + (long long)t * DDIM;
  for (int c = lane; c < DDIM; c += 32) {
    float xv = xr[c];
    #pragma unroll
    for (int e = 0; e < EEXP; ++e) acc[e] += xv * rw[e * DDIM + c];
  }
  #pragma unroll
  for (int e = 0; e < EEXP; ++e) {
    #pragma unroll
    for (int off = 16; off > 0; off >>= 1) acc[e] += __shfl_xor(acc[e], off, 32);
  }
  if (lane == 0) {
    int e0 = 0; float l0 = acc[0];
    #pragma unroll
    for (int e = 1; e < EEXP; ++e) if (acc[e] > l0) { l0 = acc[e]; e0 = e; }
    int e1 = -1; float l1 = -3.4e38f;
    #pragma unroll
    for (int e = 0; e < EEXP; ++e)
      if (e != e0 && acc[e] > l1) { l1 = acc[e]; e1 = e; }
    float w0 = 1.f / (1.f + __expf(l1 - l0));   // softmax over {l0,l1}
    float w1 = 1.f - w0;
    int p0 = atomicAdd(&cnt[e0], 1);
    tokl[e0 * N_TOK + p0] = t;                  // slot 0
    wl[e0 * N_TOK + p0]   = w0;
    int p1 = atomicAdd(&cnt[e1], 1);
    tokl[e1 * N_TOK + p1] = t | (1 << 20);      // slot 1
    wl[e1 * N_TOK + p1]   = w1;
  }
}

// ---------------------------------------------------------------- fused MoE
// Block = 256 threads (8 waves), one 32-token tile of one expert, full D out.
// Per HC=64 chunk: waves 0..7 each own one (Mtile in {0,1}, Htile in {0..3})
// gate+up 16x16 tile (24+24 WMMA over K=768), silu*up -> bf16 LDS, then every
// wave accumulates its 96-wide D slice of the down projection (12 persistent
// 16x16 f32 accumulator tiles, 24 WMMA per chunk).
__global__ __launch_bounds__(256) void _moe_gemm_kernel(
    const unsigned short* __restrict__ xb,
    const unsigned short* __restrict__ wgb,
    const unsigned short* __restrict__ wub,
    const unsigned short* __restrict__ wdb,
    const int* __restrict__ cnt,
    const int* __restrict__ tokl,
    const float* __restrict__ wl,
    float* __restrict__ contrib) {
  __shared__ __align__(16) unsigned short xs[TM * DDIM];  // 48 KB
  __shared__ __align__(16) unsigned short hs[TM * HC];    // 4 KB
  __shared__ int   s_tok[TM];
  __shared__ int   s_slot[TM];
  __shared__ float s_w[TM];

  const int e  = blockIdx.y;
  const int mt = blockIdx.x;
  const int ce = cnt[e];
  if (mt * TM >= ce) return;
  const int rows = min(TM, ce - mt * TM);

  const int tid  = threadIdx.x;
  const int lane = tid & 31;
  const int wave = tid >> 5;
  const int half = (lane < 16) ? 0 : 1;
  const int l16  = lane & 15;

  if (tid < TM) {
    if (tid < rows) {
      int entry = tokl[e * N_TOK + mt * TM + tid];
      s_tok[tid]  = entry & 0xFFFFF;
      s_slot[tid] = entry >> 20;
      s_w[tid]    = wl[e * N_TOK + mt * TM + tid];
    } else {
      s_tok[tid] = -1; s_slot[tid] = 0; s_w[tid] = 0.f;
    }
  }
  __syncthreads();

  // gather 32-token bf16 x tile into LDS via async memory->LDS copies
  for (int i = tid; i < TM * (DDIM / 8); i += blockDim.x) {
    int r = i / (DDIM / 8);
    int c = (i % (DDIM / 8)) * 8;
    int t = s_tok[r]; if (t < 0) t = 0;   // pad rows read token 0, never stored
    async_copy_b128(&xs[r * DDIM + c], &xb[(long long)t * DDIM + c]);
  }
  async_wait0();           // drain this wave's ASYNCcnt
  __syncthreads();         // make LDS visible block-wide

  const unsigned short* wg_e = wgb + (long long)e * HDIM * DDIM;
  const unsigned short* wu_e = wub + (long long)e * HDIM * DDIM;
  const unsigned short* wd_e = wdb + (long long)e * DDIM * HDIM;

  const int mt2 = wave >> 2;  // gate/up M-tile (0..1)
  const int ht  = wave & 3;   // gate/up H-tile within chunk (0..3)

  v8f dacc[12];
  {
    v8f z = {};
    #pragma unroll
    for (int i = 0; i < 12; ++i) dacc[i] = z;
  }

  for (int h0 = 0; h0 < HDIM; h0 += HC) {
    // ---- gate & up for this wave's (mt2, ht) 16x16 tile ----
    v8f gacc = {}, uacc = {};
    const int arow = mt2 * 16 + l16;
    const int hrow = h0 + ht * 16 + l16;
    const unsigned short* wgr = wg_e + (long long)hrow * DDIM;
    const unsigned short* wur = wu_e + (long long)hrow * DDIM;
    // prefetch next chunk's weight rows (global_prefetch_b8, L2-resident)
    if (h0 + HC < HDIM) {
      __builtin_prefetch(wgr + (long long)HC * DDIM, 0, 0);
      __builtin_prefetch(wur + (long long)HC * DDIM, 0, 0);
    }
    #pragma unroll 4
    for (int k0 = 0; k0 < DDIM; k0 += 32) {
      BfVec a, bg, bu;
      // A (16-bit) per-lane k order: lanes 0-15: 0..7,16..23; lanes 16-31: 8..15,24..31
      int ka0 = k0 + (half ? 8 : 0);
      int ka1 = k0 + (half ? 24 : 16);
      a.q[0] = *(const uint4*)&xs[arow * DDIM + ka0];
      a.q[1] = *(const uint4*)&xs[arow * DDIM + ka1];
      // B (32x16): same k order as A (lanes 0-15: 0..7,16..23; lanes 16-31: 8..15,24..31), col = lane%16
      int kb = k0 + (half ? 8 : 0);
      bg.q[0] = *(const uint4*)&wgr[kb];
      bg.q[1] = *(const uint4*)&wgr[kb + 16];
      bu.q[0] = *(const uint4*)&wur[kb];
      bu.q[1] = *(const uint4*)&wur[kb + 16];
      gacc = __builtin_amdgcn_wmma_f32_16x16x32_bf16(false, a.v, false, bg.v,
                                                     (short)0, gacc, false, false);
      uacc = __builtin_amdgcn_wmma_f32_16x16x32_bf16(false, a.v, false, bu.v,
                                                     (short)0, uacc, false, false);
    }
    __syncthreads();  // prior chunk's hs readers are done
    {
      // C layout: VGPR r -> row r (lanes 0-15) / row r+8 (lanes 16-31), col = lane%16
      const int col   = ht * 16 + l16;
      const int rbase = mt2 * 16 + (half ? 8 : 0);
      #pragma unroll
      for (int r = 0; r < 8; ++r) {
        float g = gacc[r];
        float u = uacc[r];
        float hv = (g / (1.f + __expf(-g))) * u;   // silu(g)*u
        hs[(rbase + r) * HC + col] = f2bf(hv);
      }
    }
    __syncthreads();
    // ---- down projection: out[m,d] += h[m,hk] * Wd[d,hk] ----
    #pragma unroll
    for (int dt = 0; dt < 6; ++dt) {
      const int drow = wave * 96 + dt * 16 + l16;
      const unsigned short* wdr = wd_e + (long long)drow * HDIM + h0;
      #pragma unroll
      for (int ka = 0; ka < HC; ka += 32) {
        BfVec b;
        int kb = ka + (half ? 8 : 0);
        b.q[0] = *(const uint4*)&wdr[kb];
        b.q[1] = *(const uint4*)&wdr[kb + 16];
        #pragma unroll
        for (int m2 = 0; m2 < 2; ++m2) {
          BfVec a;
          int hrA = m2 * 16 + l16;
          int ka0 = ka + (half ? 8 : 0);
          int ka1 = ka + (half ? 24 : 16);
          a.q[0] = *(const uint4*)&hs[hrA * HC + ka0];
          a.q[1] = *(const uint4*)&hs[hrA * HC + ka1];
          dacc[dt * 2 + m2] = __builtin_amdgcn_wmma_f32_16x16x32_bf16(
              false, a.v, false, b.v, (short)0, dacc[dt * 2 + m2], false, false);
        }
      }
    }
  }

  // epilogue: scale rows by routing weight, scatter to slot contribution plane
  #pragma unroll
  for (int dt = 0; dt < 6; ++dt) {
    const int d = wave * 96 + dt * 16 + l16;
    #pragma unroll
    for (int m2 = 0; m2 < 2; ++m2) {
      #pragma unroll
      for (int r = 0; r < 8; ++r) {
        int i = m2 * 16 + (half ? 8 : 0) + r;
        if (i < rows) {
          int t = s_tok[i];
          float v = dacc[dt * 2 + m2][r] * s_w[i];
          contrib[((long long)s_slot[i] * N_TOK + t) * DDIM + d] = v;
        }
      }
    }
  }
}

// ---------------------------------------------------------------- combine
__global__ void _moe_combine_kernel(const float* __restrict__ contrib,
                                    float* __restrict__ out) {
  const long long total = (long long)N_TOK * DDIM;
  for (long long i = blockIdx.x * (long long)blockDim.x + threadIdx.x;
       i < total; i += (long long)gridDim.x * blockDim.x) {
    out[i] = contrib[i] + contrib[total + i];
  }
}

// ---------------------------------------------------------------- launch
extern "C" void kernel_launch(void* const* d_in, const int* in_sizes, int n_in,
                              void* d_out, int out_size, void* d_ws, size_t ws_size,
                              hipStream_t stream) {
  const float* x  = (const float*)d_in[0];
  const float* rw = (const float*)d_in[1];
  const float* Wg = (const float*)d_in[2];
  const float* Wu = (const float*)d_in[3];
  const float* Wd = (const float*)d_in[4];
  float* out = (float*)d_out;

  // workspace carve-up (256B aligned); ~177 MB total
  char* ws = (char*)d_ws;
  size_t off = 0;
  auto carve = [&](size_t bytes) {
    char* p = ws + off;
    off = (off + bytes + 255) & ~(size_t)255;
    return p;
  };
  const size_t S0 = (size_t)N_TOK * DDIM;            // x elems
  const size_t S1 = (size_t)EEXP * HDIM * DDIM;      // each weight elems
  unsigned short* xb   = (unsigned short*)carve(S0 * 2);
  unsigned short* wgb  = (unsigned short*)carve(S1 * 2);
  unsigned short* wub  = (unsigned short*)carve(S1 * 2);
  unsigned short* wdb  = (unsigned short*)carve(S1 * 2);
  int*   cnt     = (int*)carve(EEXP * sizeof(int));
  int*   tokl    = (int*)carve((size_t)EEXP * N_TOK * sizeof(int));
  float* wl      = (float*)carve((size_t)EEXP * N_TOK * sizeof(float));
  float* contrib = (float*)carve((size_t)2 * N_TOK * DDIM * sizeof(float));
  (void)ws_size; (void)in_sizes; (void)n_in; (void)out_size;

  hipMemsetAsync(cnt, 0, EEXP * sizeof(int), stream);

  _moe_cvt_kernel<<<4096, 256, 0, stream>>>(x, Wg, Wu, Wd, xb, wgb, wub, wdb);

  _moe_router_kernel<<<N_TOK / 8, 256, 0, stream>>>(x, rw, cnt, tokl, wl);

  dim3 grid(N_TOK / TM, EEXP);   // worst case: one expert owns all tokens
  _moe_gemm_kernel<<<grid, 256, 0, stream>>>(xb, wgb, wub, wdb, cnt, tokl, wl,
                                             contrib);

  _moe_combine_kernel<<<2048, 256, 0, stream>>>(contrib, out);
}
